// ShiftedWindowAttnBlock_14018773254206
// MI455X (gfx1250) — matrix operand order repaired and checked by reading, not compiled, not run on hardware
//
#include <hip/hip_runtime.h>
#include <hip/hip_bf16.h>
#include <stdint.h>

// ---------------------------------------------------------------------------
// Shifted-window attention block (Swin) for gfx1250 (MI455X), wave32 + WMMA.
// GEMMs: v_wmma_f32_16x16x32_f16 with async global->LDS double buffering
// (global_load_async_to_lds_b128 / s_wait_asynccnt).
// ---------------------------------------------------------------------------

typedef __attribute__((ext_vector_type(16))) _Float16 v16h;
typedef __attribute__((ext_vector_type(8)))  float    v8f;

struct u4 { unsigned int a, b, c, d; };          // POD 16-byte chunk (LDS b128)

union FragH { v16h v; u4 u[2]; };                // 16x f16 WMMA operand fragment

__device__ inline v8f v8f_zero() {
    v8f z;
    #pragma unroll
    for (int i = 0; i < 8; ++i) z[i] = 0.0f;
    return z;
}

__device__ inline v8f wmma16x16x32(v16h a, v16h b, v8f c) {
    return __builtin_amdgcn_wmma_f32_16x16x32_f16(
        /*neg_a=*/false, a, /*neg_b=*/false, b,
        /*c_mod=*/(short)0, c, /*reuse_a=*/false, /*reuse_b=*/false);
}

// 32B (two b128) async copy global -> LDS; same INST_OFFSET applies to both
// sides, so one addr pair covers the +16B chunk. Tracked by ASYNCcnt.
__device__ inline void async_copy32(const _Float16* g, _Float16* l) {
    unsigned lds_off = (unsigned)(size_t)l;                 // low 32 bits = LDS offset
    unsigned long long ga = (unsigned long long)(size_t)g;
    asm volatile(
        "global_load_async_to_lds_b128 %0, %1, off\n\t"
        "global_load_async_to_lds_b128 %0, %1, off offset:16"
        :: "v"(lds_off), "v"(ga) : "memory");
}
__device__ inline void wait_async_le4() { asm volatile("s_wait_asynccnt 0x4" ::: "memory"); }
__device__ inline void wait_async_0()   { asm volatile("s_wait_asynccnt 0x0" ::: "memory"); }

// ---------------------------------------------------------------------------
// NCHW (8,512,56,56) -> token-major (B*H*W, C), tiled 32x32 transpose via LDS
// ---------------------------------------------------------------------------
__global__ __launch_bounds__(256)
void nchw_to_tok(const float* __restrict__ in, float* __restrict__ out) {
    __shared__ float tile[32][33];
    const int HW = 3136, C = 512;
    int hw0 = blockIdx.x * 32, c0 = blockIdx.y * 32, b = blockIdx.z;
    int tx = threadIdx.x, ty = threadIdx.y;   // 32 x 8
    #pragma unroll
    for (int i = 0; i < 32; i += 8)
        tile[ty + i][tx] = in[(size_t)(b * C + c0 + ty + i) * HW + hw0 + tx];
    __syncthreads();
    #pragma unroll
    for (int i = 0; i < 32; i += 8)
        out[(size_t)(b * HW + hw0 + ty + i) * C + c0 + tx] = tile[tx][ty + i];
}

// ---------------------------------------------------------------------------
// fp32 -> f16 elementwise (weights)
// ---------------------------------------------------------------------------
__global__ __launch_bounds__(256)
void f32tof16(const float* __restrict__ in, _Float16* __restrict__ out, int n) {
    int i = blockIdx.x * 256 + threadIdx.x;
    if (i < n) out[i] = (_Float16)in[i];
}

// ---------------------------------------------------------------------------
// LayerNorm over C=512: one block per token, optional fp32 + f16 outputs
// ---------------------------------------------------------------------------
__global__ __launch_bounds__(256)
void layernorm_k(const float* __restrict__ x, const float* __restrict__ g,
                 const float* __restrict__ bb, float* __restrict__ of32,
                 _Float16* __restrict__ of16) {
    __shared__ float s1[256], s2[256];
    const int C = 512;
    int t = blockIdx.x, tid = threadIdx.x;
    const float* row = x + (size_t)t * C;
    float a0 = row[tid], a1 = row[tid + 256];
    s1[tid] = a0 + a1;
    s2[tid] = a0 * a0 + a1 * a1;
    __syncthreads();
    #pragma unroll
    for (int s = 128; s > 0; s >>= 1) {
        if (tid < s) { s1[tid] += s1[tid + s]; s2[tid] += s2[tid + s]; }
        __syncthreads();
    }
    float mean = s1[0] * (1.0f / 512.0f);
    float var  = s2[0] * (1.0f / 512.0f) - mean * mean;
    float rstd = rsqrtf(var + 1e-5f);
    float v0 = (a0 - mean) * rstd * g[tid]       + bb[tid];
    float v1 = (a1 - mean) * rstd * g[tid + 256] + bb[tid + 256];
    if (of32) {
        of32[(size_t)t * C + tid]       = v0;
        of32[(size_t)t * C + tid + 256] = v1;
    }
    if (of16) {
        of16[(size_t)t * C + tid]       = (_Float16)v0;
        of16[(size_t)t * C + tid + 256] = (_Float16)v1;
    }
}

// ---------------------------------------------------------------------------
// WMMA GEMM: C[M,N] = A[M,K] * W[N,K]^T (+bias, q-scale, GELU, residual)
// Block tile 128x128, BK=32; 8 waves, each wave 32x64 (2x4 fragments).
// Async double-buffered global->LDS tile pipeline (ASYNCcnt).
// ---------------------------------------------------------------------------
#define GBM 128
#define GBN 128
#define GBK 32
#define GLDS 40   // padded half-stride (80B, multiple of 16B, bank-friendly)

__global__ __launch_bounds__(256)
void gemm_f16_wmma(const _Float16* __restrict__ A, const _Float16* __restrict__ W,
                   const float* __restrict__ bias, float* __restrict__ outF32,
                   _Float16* __restrict__ outF16, const float* __restrict__ res,
                   int M, int N, int K, int gelu, float qscale, int scale_ncols) {
    __shared__ _Float16 As[2][GBM * GLDS];
    __shared__ _Float16 Bs[2][GBN * GLDS];

    int tid  = threadIdx.x;
    int lane = tid & 31;
    int wave = tid >> 5;
    int wm = wave >> 1;          // 0..3  (32 rows each)
    int wn = wave &  1;          // 0..1  (64 cols each)
    int l16 = lane & 15;
    int lh  = lane >> 4;
    int bm = blockIdx.y * GBM;
    int bn = blockIdx.x * GBN;

    v8f acc[2][4];
    #pragma unroll
    for (int i = 0; i < 2; ++i)
        #pragma unroll
        for (int j = 0; j < 4; ++j) acc[i][j] = v8f_zero();

    int row = tid >> 1;          // 0..127
    int seg = tid & 1;           // 16-half segment within a BK row

    const _Float16* gA = A + (size_t)(bm + row) * K + seg * 16;
    const _Float16* gB = W + (size_t)(bn + row) * K + seg * 16;
    int ldst = row * GLDS + seg * 16;

    int nk = K / GBK;
    // prefetch tile 0 into buffer 0 (4 async b128 per thread per tile pair)
    async_copy32(gA, &As[0][ldst]);
    async_copy32(gB, &Bs[0][ldst]);

    int buf = 0;
    for (int s = 0; s < nk; ++s) {
        if (s + 1 < nk) {
            async_copy32(gA + (s + 1) * GBK, &As[buf ^ 1][ldst]);
            async_copy32(gB + (s + 1) * GBK, &Bs[buf ^ 1][ldst]);
            wait_async_le4();    // in-order: tile s resident, prefetch in flight
        } else {
            wait_async_0();
        }
        __syncthreads();

        FragH afr[2];
        #pragma unroll
        for (int mt = 0; mt < 2; ++mt) {
            const _Float16* base = &As[buf][(wm * 32 + mt * 16 + l16) * GLDS];
            afr[mt].u[0] = *(const u4*)(base + lh * 8);        // K 0-7 / 8-15
            afr[mt].u[1] = *(const u4*)(base + 16 + lh * 8);   // K 16-23 / 24-31
        }
        #pragma unroll
        for (int nt = 0; nt < 4; ++nt) {
            FragH bfr;
            const _Float16* base = &Bs[buf][(wn * 64 + nt * 16 + l16) * GLDS];
            bfr.u[0] = *(const u4*)(base + lh * 8);
            bfr.u[1] = *(const u4*)(base + 16 + lh * 8);
            #pragma unroll
            for (int mt = 0; mt < 2; ++mt)
                acc[mt][nt] = wmma16x16x32(afr[mt].v, bfr.v, acc[mt][nt]);
        }
        __syncthreads();         // reads of buf done before it is refilled
        buf ^= 1;
    }

    // epilogue: D layout — lane holds column n; VGPR r -> row r (+8 for lanes 16-31)
    #pragma unroll
    for (int mt = 0; mt < 2; ++mt)
        #pragma unroll
        for (int nt = 0; nt < 4; ++nt) {
            int n = bn + wn * 64 + nt * 16 + l16;
            float bv = bias ? bias[n] : 0.0f;
            float sc = (n < scale_ncols) ? qscale : 1.0f;
            #pragma unroll
            for (int r = 0; r < 8; ++r) {
                int m = bm + wm * 32 + mt * 16 + lh * 8 + r;
                float v = (acc[mt][nt][r] + bv) * sc;
                if (gelu) v = 0.5f * v * (1.0f + erff(v * 0.70710678118654752f));
                if (res)    v += res[(size_t)m * N + n];
                if (outF32) outF32[(size_t)m * N + n] = v;
                if (outF16) outF16[(size_t)m * N + n] = (_Float16)v;
            }
        }
}

// ---------------------------------------------------------------------------
// Windowed attention: one block per (window, head). 4 waves, 128 threads.
// Scores: 16 WMMA (K=32=hd).  P*V: 16 WMMA (K=64 padded tokens).
// ---------------------------------------------------------------------------
#define AQS 40     // Q/K row stride (halfs)
#define AVS 72     // Vt / P row stride (halfs)
#define ASS 65     // score row stride (floats)

__global__ __launch_bounds__(128)
void win_attn(const _Float16* __restrict__ qkv, const float* __restrict__ table,
              _Float16* __restrict__ out, int shift, int use_mask) {
    __shared__ _Float16 Qs[64 * AQS];
    __shared__ _Float16 Ks[64 * AQS];
    __shared__ _Float16 Vt[32 * AVS];   // transposed: [dim][token]
    __shared__ float    Sb[64 * ASS];
    __shared__ _Float16 Pb[64 * AVS];

    int wi   = blockIdx.x;      // 0..511 = b*64 + wy*8 + wx
    int head = blockIdx.y;      // 0..15
    int b  = wi >> 6;
    int wy = (wi >> 3) & 7;
    int wx = wi & 7;
    int tid  = threadIdx.x;
    int lane = tid & 31;
    int wv   = tid >> 5;        // wave id: 16-row strip
    int l16 = lane & 15;
    int lh  = lane >> 4;

    // ---- gather Q/K/V head slices into LDS (zero-padded to 64 tokens) ----
    for (int j = tid; j < 192; j += 128) {
        int tok = j / 3, which = j % 3;
        union { u4 u[4]; _Float16 h[32]; } buf;
        if (tok < 49) {
            int yl = tok / 7, xl = tok % 7;
            int y = (wy * 7 + yl + shift) % 56;   // roll(x, -shift) source pixel
            int x = (wx * 7 + xl + shift) % 56;
            int g = b * 3136 + y * 56 + x;
            const u4* src = (const u4*)(qkv + (size_t)g * 1536 + which * 512 + head * 32);
            #pragma unroll
            for (int i = 0; i < 4; ++i) buf.u[i] = src[i];
        } else {
            #pragma unroll
            for (int i = 0; i < 4; ++i) buf.u[i] = u4{0u, 0u, 0u, 0u};
        }
        if (which == 0) {
            u4* d = (u4*)(Qs + tok * AQS);
            #pragma unroll
            for (int i = 0; i < 4; ++i) d[i] = buf.u[i];
        } else if (which == 1) {
            u4* d = (u4*)(Ks + tok * AQS);
            #pragma unroll
            for (int i = 0; i < 4; ++i) d[i] = buf.u[i];
        } else {
            #pragma unroll
            for (int d = 0; d < 32; ++d) Vt[d * AVS + tok] = buf.h[d];
        }
    }
    __syncthreads();

    // ---- scores: S[m,n] = sum_k Q[m,k]*K[n,k]  (q already scaled) ----
    FragH aq;
    {
        const _Float16* base = Qs + (wv * 16 + l16) * AQS;
        aq.u[0] = *(const u4*)(base + lh * 8);
        aq.u[1] = *(const u4*)(base + 16 + lh * 8);
    }
    #pragma unroll
    for (int nt = 0; nt < 4; ++nt) {
        FragH bk;
        const _Float16* base = Ks + (nt * 16 + l16) * AQS;
        bk.u[0] = *(const u4*)(base + lh * 8);
        bk.u[1] = *(const u4*)(base + 16 + lh * 8);
        v8f s = wmma16x16x32(aq.v, bk.v, v8f_zero());
        #pragma unroll
        for (int r = 0; r < 8; ++r) {
            int m = wv * 16 + lh * 8 + r;
            int n = nt * 16 + l16;
            float val = -1e30f;
            if (m < 49 && n < 49) {
                int my = m / 7, mx = m % 7, ny = n / 7, nx = n % 7;
                int idx = (my - ny + 6) * 13 + (mx - nx + 6);
                val = s[r] + table[idx * 16 + head];
                if (use_mask) {
                    int Ym = wy * 7 + my, Xm = wx * 7 + mx;
                    int Yn = wy * 7 + ny, Xn = wx * 7 + nx;
                    int gm = (Ym < 49 ? 0 : (Ym < 53 ? 1 : 2)) * 3 +
                             (Xm < 49 ? 0 : (Xm < 53 ? 1 : 2));
                    int gn = (Yn < 49 ? 0 : (Yn < 53 ? 1 : 2)) * 3 +
                             (Xn < 49 ? 0 : (Xn < 53 ? 1 : 2));
                    if (gm != gn) val -= 100.0f;
                }
            }
            Sb[m * ASS + n] = val;
        }
    }
    __syncthreads();

    // ---- softmax per row (padding cols go to zero via -1e30) ----
    if (tid < 64) {
        int m = tid;
        float mx = -1e30f;
        for (int n = 0; n < 64; ++n) mx = fmaxf(mx, Sb[m * ASS + n]);
        float sum = 0.0f;
        for (int n = 0; n < 64; ++n) sum += expf(Sb[m * ASS + n] - mx);
        float rs = 1.0f / sum;
        for (int n = 0; n < 64; ++n)
            Pb[m * AVS + n] = (_Float16)(expf(Sb[m * ASS + n] - mx) * rs);
    }
    __syncthreads();

    // ---- out[m,d] = sum_n P[m,n] * V[n,d]  via Vt ----
    v8f o[2];
    o[0] = v8f_zero(); o[1] = v8f_zero();
    #pragma unroll
    for (int ks = 0; ks < 2; ++ks) {
        FragH ap;
        const _Float16* base = Pb + (wv * 16 + l16) * AVS + ks * 32;
        ap.u[0] = *(const u4*)(base + lh * 8);
        ap.u[1] = *(const u4*)(base + 16 + lh * 8);
        #pragma unroll
        for (int nt = 0; nt < 2; ++nt) {
            FragH bv;
            const _Float16* vb = Vt + (nt * 16 + l16) * AVS + ks * 32;
            bv.u[0] = *(const u4*)(vb + lh * 8);
            bv.u[1] = *(const u4*)(vb + 16 + lh * 8);
            o[nt] = wmma16x16x32(ap.v, bv.v, o[nt]);
        }
    }
    #pragma unroll
    for (int nt = 0; nt < 2; ++nt) {
        int n = nt * 16 + l16;                 // head-dim 0..31
        #pragma unroll
        for (int r = 0; r < 8; ++r) {
            int m = wv * 16 + lh * 8 + r;
            if (m < 49)
                out[(size_t)(wi * 49 + m) * 512 + head * 32 + n] = (_Float16)o[nt][r];
        }
    }
}

// ---------------------------------------------------------------------------
// out[tok] = base[tok] + src[window_token(tok)]  (un-partition + un-roll)
// ---------------------------------------------------------------------------
__global__ __launch_bounds__(256)
void add_scatter(const float* __restrict__ base, const float* __restrict__ src,
                 float* __restrict__ out, int shift) {
    int idx = blockIdx.x * 256 + threadIdx.x;   // over T*C = 12.85M
    int t = idx >> 9;
    int c = idx & 511;
    int b = t / 3136, rem = t % 3136;
    int y = rem / 56, x = rem % 56;
    int Y = (y - shift + 56) % 56;
    int X = (x - shift + 56) % 56;
    int wt = (b * 64 + (Y / 7) * 8 + (X / 7)) * 49 + (Y % 7) * 7 + (X % 7);
    out[idx] = base[idx] + src[(size_t)wt * 512 + c];
}

// ---------------------------------------------------------------------------
// Orchestration
// ---------------------------------------------------------------------------
extern "C" void kernel_launch(void* const* d_in, const int* in_sizes, int n_in,
                              void* d_out, int out_size, void* d_ws, size_t ws_size,
                              hipStream_t stream) {
    const int T = 25088, C = 512, C3 = 1536, Cm = 2048;
    const size_t TC = (size_t)T * C;

    char* ws = (char*)d_ws;
    size_t off = 0;
    auto wsalloc = [&](size_t bytes) -> void* {
        void* p = ws + off;
        off = (off + bytes + 255) & ~(size_t)255;
        return p;
    };
    // ~450 MB total workspace
    float*    X     = (float*)wsalloc(TC * 4);
    float*    XN    = (float*)wsalloc(TC * 4);
    float*    X2    = (float*)wsalloc(TC * 4);
    float*    TMP   = (float*)wsalloc(TC * 4);
    _Float16* A16   = (_Float16*)wsalloc(TC * 2);
    _Float16* QKV16 = (_Float16*)wsalloc((size_t)T * C3 * 2);
    _Float16* ATT16 = (_Float16*)wsalloc(TC * 2);
    _Float16* M16   = (_Float16*)wsalloc((size_t)T * Cm * 2);
    _Float16 *Wqkv16[2], *Wproj16[2], *Wfc116[2], *Wfc216[2];
    for (int p = 0; p < 2; ++p) {
        Wqkv16[p]  = (_Float16*)wsalloc((size_t)C3 * C * 2);
        Wproj16[p] = (_Float16*)wsalloc((size_t)C * C * 2);
        Wfc116[p]  = (_Float16*)wsalloc((size_t)Cm * C * 2);
        Wfc216[p]  = (_Float16*)wsalloc((size_t)C * Cm * 2);
    }

    nchw_to_tok<<<dim3(98, 16, 8), dim3(32, 8), 0, stream>>>((const float*)d_in[0], X);

    for (int p = 0; p < 2; ++p) {
        int base = 1 + p * 13;
        int nq = C3 * C, np = C * C, n1 = Cm * C, n2 = C * Cm;
        f32tof16<<<(nq + 255) / 256, 256, 0, stream>>>((const float*)d_in[base + 2],  Wqkv16[p],  nq);
        f32tof16<<<(np + 255) / 256, 256, 0, stream>>>((const float*)d_in[base + 4],  Wproj16[p], np);
        f32tof16<<<(n1 + 255) / 256, 256, 0, stream>>>((const float*)d_in[base + 9],  Wfc116[p],  n1);
        f32tof16<<<(n2 + 255) / 256, 256, 0, stream>>>((const float*)d_in[base + 11], Wfc216[p],  n2);
    }

    const float qscale = 0.17677669529663689f;  // 1/sqrt(hd=32)
    float* cur = X;
    for (int p = 0; p < 2; ++p) {
        int base = 1 + p * 13;
        const float* n1g   = (const float*)d_in[base + 0];
        const float* n1b   = (const float*)d_in[base + 1];
        const float* qkvb  = (const float*)d_in[base + 3];
        const float* projb = (const float*)d_in[base + 5];
        const float* table = (const float*)d_in[base + 6];
        const float* n2g   = (const float*)d_in[base + 7];
        const float* n2b   = (const float*)d_in[base + 8];
        const float* fc1b  = (const float*)d_in[base + 10];
        const float* fc2b  = (const float*)d_in[base + 12];
        int shift = p ? 3 : 0;

        // LN1: fp32 kept (sw-branch residual uses xn), f16 for GEMM
        layernorm_k<<<T, 256, 0, stream>>>(cur, n1g, n1b, XN, A16);
        // QKV GEMM (+bias, q columns scaled), f16 out
        gemm_f16_wmma<<<dim3(C3 / 128, T / 128), 256, 0, stream>>>(
            A16, Wqkv16[p], qkvb, nullptr, QKV16, nullptr, T, C3, C, 0, qscale, C);
        // attention (window-ordered f16 out)
        win_attn<<<dim3(512, 16), 128, 0, stream>>>(QKV16, table, ATT16, shift, p ? 1 : 0);
        // proj GEMM, fp32 out (window order)
        gemm_f16_wmma<<<dim3(C / 128, T / 128), 256, 0, stream>>>(
            ATT16, Wproj16[p], projb, TMP, nullptr, nullptr, T, C, C, 0, 1.0f, 0);
        // residual + un-partition (+un-roll): base = shortcut (p=0) or xn (p=1)
        add_scatter<<<(int)(TC / 256), 256, 0, stream>>>(p == 0 ? cur : XN, TMP, X2, shift);
        // LN2 -> f16
        layernorm_k<<<T, 256, 0, stream>>>(X2, n2g, n2b, nullptr, A16);
        // fc1 GEMM + exact GELU, f16 out
        gemm_f16_wmma<<<dim3(Cm / 128, T / 128), 256, 0, stream>>>(
            A16, Wfc116[p], fc1b, nullptr, M16, nullptr, T, Cm, C, 1, 1.0f, 0);
        // fc2 GEMM + residual; final branch writes fp32 straight into d_out
        float* outp = (p == 0) ? X : (float*)d_out;
        gemm_f16_wmma<<<dim3(C / 128, T / 128), 256, 0, stream>>>(
            M16, Wfc216[p], fc2b, outp, nullptr, X2, T, C, Cm, 0, 1.0f, 0);
        cur = X;
    }
}
